// CTMHead_29540785062024
// MI455X (gfx1250) — matrix operand order repaired and checked by reading, not compiled, not run on hardware
//
#include <hip/hip_runtime.h>

#define HID_   512
#define NN_    256
#define TICKS_ 10
#define HIST_  20
#define VOCAB_ 151936
#define NLMH_  64
#define B_     256
#define XS_STRIDE 40   // padded LDS row stride (bf16 elems): 80B, 16B-aligned, bank-spread

typedef __bf16 bf16;
typedef __attribute__((ext_vector_type(16))) __bf16 v16bf;
typedef __attribute__((ext_vector_type(8)))  __bf16 bf16x8;
typedef __attribute__((ext_vector_type(8)))  float   v8f;
typedef __attribute__((ext_vector_type(4)))  float   f32x4;

__device__ __forceinline__ float gelu_exact(float x) {
  return 0.5f * x * (1.0f + erff(x * 0.70710678118654752f));
}

// ---------------- BitNet gamma (deterministic two-stage) ----------------
__global__ void __launch_bounds__(256)
partial_absclip_kernel(const float* __restrict__ w, int count, float* __restrict__ partials) {
  __shared__ float red[256];
  float s = 0.f;
  for (int i = blockIdx.x * blockDim.x + threadIdx.x; i < count; i += gridDim.x * blockDim.x) {
    float v = fminf(fmaxf(w[i], -2.f), 2.f);
    s += fabsf(v);
  }
  red[threadIdx.x] = s; __syncthreads();
  for (int off = 128; off > 0; off >>= 1) {
    if (threadIdx.x < off) red[threadIdx.x] += red[threadIdx.x + off];
    __syncthreads();
  }
  if (threadIdx.x == 0) partials[blockIdx.x] = red[0];
}

__global__ void __launch_bounds__(256)
finalize_gamma_kernel(const float* __restrict__ partials, int count, float* __restrict__ gamma_out) {
  __shared__ float red[256];
  red[threadIdx.x] = partials[threadIdx.x];
  __syncthreads();
  for (int off = 128; off > 0; off >>= 1) {
    if (threadIdx.x < off) red[threadIdx.x] += red[threadIdx.x + off];
    __syncthreads();
  }
  if (threadIdx.x == 0) gamma_out[0] = fmaxf(red[0] / (float)count, 1e-4f);
}

__global__ void __launch_bounds__(256)
quantize_bf16_kernel(const float* __restrict__ w, int count,
                     const float* __restrict__ gamma_p, bf16* __restrict__ out) {
  float gamma = gamma_p[0];
  for (int i = blockIdx.x * blockDim.x + threadIdx.x; i < count; i += gridDim.x * blockDim.x) {
    float v = fminf(fmaxf(w[i], -2.f), 2.f);
    float q = rintf(v / gamma);
    q = fminf(fmaxf(q, -1.f), 1.f);
    out[i] = (bf16)(q * gamma);
  }
}

__global__ void __launch_bounds__(256)
cvt_bf16_kernel(const float* __restrict__ w, int count, bf16* __restrict__ out) {
  for (int i = blockIdx.x * blockDim.x + threadIdx.x; i < count; i += gridDim.x * blockDim.x)
    out[i] = (bf16)w[i];
}

__global__ void __launch_bounds__(256)
pack_fused_kernel(const float* __restrict__ l0, const float* __restrict__ l1, bf16* __restrict__ out) {
  for (int i = blockIdx.x * blockDim.x + threadIdx.x; i < B_ * 2 * HID_; i += gridDim.x * blockDim.x) {
    int b = i / (2 * HID_), j = i % (2 * HID_);
    float v = (j < HID_) ? l0[b * HID_ + j] : l1[b * HID_ + j - HID_];
    out[i] = (bf16)v;
  }
}

__global__ void __launch_bounds__(256)
init_z_kernel(const float* __restrict__ init_z, bf16* __restrict__ zbf) {
  for (int i = blockIdx.x * blockDim.x + threadIdx.x; i < B_ * NN_; i += gridDim.x * blockDim.x)
    zbf[i] = (bf16)init_z[i % NN_];
}

__global__ void __launch_bounds__(256)
init_hist_kernel(const float* __restrict__ init_hist, float* __restrict__ hist) {
  for (int i = blockIdx.x * blockDim.x + threadIdx.x; i < B_ * NN_ * HIST_; i += gridDim.x * blockDim.x)
    hist[i] = init_hist[i % (NN_ * HIST_)];
}

// ---------------- WMMA GEMM (small, L2-resident): out[M,N] = X @ W^T + bias ----------------
__global__ void __launch_bounds__(128)
gemm_bf16_kernel(const bf16* __restrict__ X, const bf16* __restrict__ W,
                 const float* __restrict__ bias, float* __restrict__ out,
                 int N, int K) {
  int lane = threadIdx.x & 31;
  int wid  = threadIdx.x >> 5;
  int nt   = blockIdx.x * 4 + wid;
  int mt   = blockIdx.y;
  int half = lane >> 4;
  int l15  = lane & 15;

  const bf16* xrow = X + (size_t)(mt * 16 + l15) * K + half * 16;
  const bf16* wrow = W + (size_t)(nt * 16 + l15) * K + half * 16;

  v8f c = {};
  for (int k0 = 0; k0 < K; k0 += 32) {
    bf16x8 alo = *(const bf16x8*)(xrow + k0);
    bf16x8 ahi = *(const bf16x8*)(xrow + k0 + 8);
    v16bf a = __builtin_shufflevector(alo, ahi, 0,1,2,3,4,5,6,7,8,9,10,11,12,13,14,15);
    v16bf b = *(const v16bf*)(wrow + k0);
    c = __builtin_amdgcn_wmma_f32_16x16x32_bf16(false, a, false, b, (short)0, c, false, false);
  }
  float bv = bias ? bias[nt * 16 + l15] : 0.f;
  size_t outbase = (size_t)(mt * 16) * N + (size_t)nt * 16 + l15;
#pragma unroll
  for (int v = 0; v < 8; v++) {
    int m = v + half * 8;
    out[outbase + (size_t)m * N] = c[v] + bv;
  }
}

// ---------------- Vocab GEMM: logits[256,N] = X[256,512]bf16 @ W[N,512]f32^T + bias ----------
// 8 waves/block; wave w owns N-tile nt=blockIdx.x*8+w and ALL 16 M-tiles (128 acc VGPRs),
// so each f32 weight is streamed from HBM exactly once (non-temporal).
// X k-slab staged to LDS once per block per k-step via GLOBAL_LOAD_ASYNC_TO_LDS_B128
// (ASYNCcnt-tracked, drained with s_wait_asynccnt before the workgroup barrier).
__global__ void __launch_bounds__(256)
gemm_vocab_kernel(const bf16* __restrict__ X, const float* __restrict__ W,
                  const float* __restrict__ bias, float* __restrict__ out,
                  int N, int K) {
  __shared__ bf16 Xs[256 * XS_STRIDE];
  int t    = threadIdx.x;
  int lane = t & 31;
  int wid  = t >> 5;
  int nt   = blockIdx.x * 8 + wid;
  int half = lane >> 4;
  int l15  = lane & 15;

  const float* wrow = W + (size_t)(nt * 16 + l15) * K + half * 16;
  const bf16*  xsrc = X + (size_t)t * K;                        // thread t stages batch-row t
  unsigned lds_dst  = (unsigned)(size_t)&Xs[t * XS_STRIDE];     // low 32 bits = LDS byte offset

  v8f acc[16];
#pragma unroll
  for (int i = 0; i < 16; i++) acc[i] = (v8f){};

  for (int k0 = 0; k0 < K; k0 += 32) {
    if (k0) __syncthreads();  // previous slab fully consumed
    // async-stage X[:, k0:k0+32] (64B per thread, 4x B128)
    asm volatile("global_load_async_to_lds_b128 %0, %1, off"
                 :: "v"(lds_dst),      "v"((const void*)(xsrc + k0))      : "memory");
    asm volatile("global_load_async_to_lds_b128 %0, %1, off"
                 :: "v"(lds_dst + 16), "v"((const void*)(xsrc + k0 + 8))  : "memory");
    asm volatile("global_load_async_to_lds_b128 %0, %1, off"
                 :: "v"(lds_dst + 32), "v"((const void*)(xsrc + k0 + 16)) : "memory");
    asm volatile("global_load_async_to_lds_b128 %0, %1, off"
                 :: "v"(lds_dst + 48), "v"((const void*)(xsrc + k0 + 24)) : "memory");

    // stream the weight slab (single-use -> non-temporal) while the async copy flies
    if (k0 + 256 < K) __builtin_prefetch(wrow + k0 + 256, 0, 0);
    float wf[16];
    *(f32x4*)(wf + 0)  = __builtin_nontemporal_load((const f32x4*)(wrow + k0 + 0));
    *(f32x4*)(wf + 4)  = __builtin_nontemporal_load((const f32x4*)(wrow + k0 + 4));
    *(f32x4*)(wf + 8)  = __builtin_nontemporal_load((const f32x4*)(wrow + k0 + 8));
    *(f32x4*)(wf + 12) = __builtin_nontemporal_load((const f32x4*)(wrow + k0 + 12));
    v16bf b;
#pragma unroll
    for (int i = 0; i < 16; i++) b[i] = (bf16)wf[i];

    asm volatile("s_wait_asynccnt 0x0" ::: "memory");
    __syncthreads();

#pragma unroll
    for (int mt = 0; mt < 16; mt++) {
      const bf16* ap = &Xs[(mt * 16 + l15) * XS_STRIDE + half * 16];
      bf16x8 alo = *(const bf16x8*)ap;
      bf16x8 ahi = *(const bf16x8*)(ap + 8);
      v16bf a = __builtin_shufflevector(alo, ahi, 0,1,2,3,4,5,6,7,8,9,10,11,12,13,14,15);
      acc[mt] = __builtin_amdgcn_wmma_f32_16x16x32_bf16(false, a, false, b, (short)0, acc[mt], false, false);
    }
  }

  float bv = bias[nt * 16 + l15];
  size_t colbase = (size_t)nt * 16 + l15;
#pragma unroll
  for (int mt = 0; mt < 16; mt++) {
#pragma unroll
    for (int v = 0; v < 8; v++) {
      int m = mt * 16 + v + half * 8;
      __builtin_nontemporal_store(acc[mt][v] + bv, out + (size_t)m * N + colbase);
    }
  }
}

// ---------------- LayerNorm (+optional exact GELU), row length 512 ----------------
__global__ void __launch_bounds__(256)
ln_kernel(const float* __restrict__ in, const float* __restrict__ g, const float* __restrict__ bta,
          int apply_gelu,
          float* __restrict__ of32, int of32_stride,
          bf16* __restrict__ obf, int obf_stride) {
  int row = blockIdx.x;
  int t = threadIdx.x;
  const float* x = in + (size_t)row * HID_;
  float x0 = x[t], x1 = x[t + 256];
  __shared__ float s1[256], s2[256];
  s1[t] = x0 + x1;
  s2[t] = x0 * x0 + x1 * x1;
  __syncthreads();
  for (int off = 128; off > 0; off >>= 1) {
    if (t < off) { s1[t] += s1[t + off]; s2[t] += s2[t + off]; }
    __syncthreads();
  }
  float mean = s1[0] * (1.f / HID_);
  float var  = s2[0] * (1.f / HID_) - mean * mean;
  float inv  = rsqrtf(var + 1e-5f);
#pragma unroll
  for (int p = 0; p < 2; p++) {
    int j = t + p * 256;
    float xv = (p == 0) ? x0 : x1;
    float y = (xv - mean) * inv * g[j] + bta[j];
    if (apply_gelu) y = gelu_exact(y);
    if (of32) of32[(size_t)row * of32_stride + j] = y;
    if (obf)  obf[(size_t)row * obf_stride + j] = (bf16)y;
  }
}

// ---------------- history shift + neuron-level MLP ----------------
__global__ void __launch_bounds__(256)
nlm_kernel(const float* __restrict__ pre, float* __restrict__ hist,
           const float* __restrict__ n1w, const float* __restrict__ n1b,
           const float* __restrict__ n2w, const float* __restrict__ n2b,
           float* __restrict__ Aacts, int tick, bf16* __restrict__ zbf) {
  __shared__ float w1[NLMH_ * HIST_];
  __shared__ float b1[NLMH_];
  __shared__ float w2[NLMH_];
  int t = threadIdx.x;
  for (int i = t; i < NLMH_ * HIST_; i += 256) w1[i] = n1w[i];
  if (t < NLMH_) { b1[t] = n1b[t]; w2[t] = n2w[t]; }
  __syncthreads();
  int b = blockIdx.x, n = t;
  float* h = hist + ((size_t)b * NN_ + n) * HIST_;
  float hv[HIST_];
#pragma unroll
  for (int i = 0; i < HIST_ - 1; i++) hv[i] = h[i + 1];
  hv[HIST_ - 1] = pre[(size_t)b * NN_ + n];
#pragma unroll
  for (int i = 0; i < HIST_; i++) h[i] = hv[i];
  float acc = n2b[0];
  for (int j = 0; j < NLMH_; j++) {
    float u = b1[j];
#pragma unroll
    for (int i = 0; i < HIST_; i++) u += hv[i] * w1[j * HIST_ + i];
    acc += gelu_exact(u) * w2[j];
  }
  if (acc != acc) acc = 0.f;
  else acc = fminf(fmaxf(acc, -5.f), 5.f);
  Aacts[((size_t)b * TICKS_ + tick) * NN_ + n] = acc;
  zbf[(size_t)b * NN_ + n] = (bf16)acc;
}

// ---------------- tick aggregation: mean -> z_agg, unbiased var -> sync ----------------
__global__ void __launch_bounds__(256)
agg_kernel(const float* __restrict__ Aacts, bf16* __restrict__ zaggbf, float* __restrict__ sync_out) {
  int b = blockIdx.x, n = threadIdx.x;
  const float* a = Aacts + (size_t)b * TICKS_ * NN_ + n;
  float vals[TICKS_];
  float s = 0.f;
#pragma unroll
  for (int t = 0; t < TICKS_; t++) { vals[t] = a[(size_t)t * NN_]; s += vals[t]; }
  float mean = s * (1.f / TICKS_);
  float ss = 0.f;
#pragma unroll
  for (int t = 0; t < TICKS_; t++) { float d = vals[t] - mean; ss += d * d; }
  float var = ss * (1.f / (TICKS_ - 1));
  zaggbf[(size_t)b * NN_ + n] = (bf16)mean;
  __shared__ float red[256];
  red[n] = var; __syncthreads();
  for (int off = 128; off > 0; off >>= 1) {
    if (n < off) red[n] += red[n + off];
    __syncthreads();
  }
  if (n == 0) sync_out[b] = red[0] * (1.f / NN_);
}

// ---------------- certainty head ----------------
__global__ void __launch_bounds__(256)
cert_kernel(const float* __restrict__ zproj, const float* __restrict__ cw,
            const float* __restrict__ cb, float* __restrict__ out) {
  int b = blockIdx.x, t = threadIdx.x;
  const float* x = zproj + (size_t)b * HID_;
  float s = x[t] * cw[t] + x[t + 256] * cw[t + 256];
  __shared__ float red[256];
  red[t] = s; __syncthreads();
  for (int off = 128; off > 0; off >>= 1) {
    if (t < off) red[t] += red[t + off];
    __syncthreads();
  }
  if (t == 0) out[b] = 1.f / (1.f + expf(-(red[0] + cb[0])));
}

extern "C" void kernel_launch(void* const* d_in, const int* in_sizes, int n_in,
                              void* d_out, int out_size, void* d_ws, size_t ws_size,
                              hipStream_t stream) {
  const float* l0      = (const float*)d_in[0];
  const float* l1      = (const float*)d_in[1];
  const float* ip_w    = (const float*)d_in[2];
  const float* ip_b    = (const float*)d_in[3];
  const float* ip_ln_g = (const float*)d_in[4];
  const float* ip_ln_b = (const float*)d_in[5];
  const float* zp_w    = (const float*)d_in[6];
  const float* zp_b    = (const float*)d_in[7];
  const float* zn_g    = (const float*)d_in[8];
  const float* zn_b    = (const float*)d_in[9];
  const float* s1_w    = (const float*)d_in[10];
  const float* s1_b    = (const float*)d_in[11];
  const float* ln1_g   = (const float*)d_in[12];
  const float* ln1_b   = (const float*)d_in[13];
  const float* s2_w    = (const float*)d_in[14];
  const float* s2_b    = (const float*)d_in[15];
  const float* ln2_g   = (const float*)d_in[16];
  const float* ln2_b   = (const float*)d_in[17];
  const float* s3_w    = (const float*)d_in[18];
  const float* s3_b    = (const float*)d_in[19];
  const float* n1_w    = (const float*)d_in[20];
  const float* n1_b    = (const float*)d_in[21];
  const float* n2_w    = (const float*)d_in[22];
  const float* n2_b    = (const float*)d_in[23];
  const float* np_w    = (const float*)d_in[24];
  const float* np_b    = (const float*)d_in[25];
  const float* np_ln_g = (const float*)d_in[26];
  const float* np_ln_b = (const float*)d_in[27];
  const float* out_w   = (const float*)d_in[28];
  const float* out_b   = (const float*)d_in[29];
  const float* cert_w  = (const float*)d_in[30];
  const float* cert_b  = (const float*)d_in[31];
  const float* init_z  = (const float*)d_in[32];
  const float* init_h  = (const float*)d_in[33];
  (void)in_sizes; (void)n_in; (void)out_size; (void)ws_size;

  size_t off = 0;
  auto alloc = [&](size_t bytes) -> void* {
    void* p = (char*)d_ws + off;
    off += (bytes + 255) & ~(size_t)255;
    return p;
  };
  float* gam      = (float*)alloc(8 * sizeof(float));
  float* partials = (float*)alloc(256 * sizeof(float));
  bf16*  ipq      = (bf16*)alloc((size_t)HID_ * 2 * HID_ * 2);
  bf16*  s1q      = (bf16*)alloc((size_t)HID_ * 2 * HID_ * 2);
  bf16*  s2q      = (bf16*)alloc((size_t)HID_ * HID_ * 2);
  bf16*  npq      = (bf16*)alloc((size_t)HID_ * NN_ * 2);
  bf16*  zpwb     = (bf16*)alloc((size_t)HID_ * NN_ * 2);
  bf16*  s3wb     = (bf16*)alloc((size_t)NN_ * HID_ * 2);
  bf16*  fusedb   = (bf16*)alloc((size_t)B_ * 2 * HID_ * 2);
  bf16*  hcatb    = (bf16*)alloc((size_t)B_ * 2 * HID_ * 2);
  bf16*  h1b      = (bf16*)alloc((size_t)B_ * HID_ * 2);
  bf16*  h2b      = (bf16*)alloc((size_t)B_ * HID_ * 2);
  bf16*  zbf      = (bf16*)alloc((size_t)B_ * NN_ * 2);
  bf16*  zaggb    = (bf16*)alloc((size_t)B_ * NN_ * 2);
  bf16*  zprojb   = (bf16*)alloc((size_t)B_ * HID_ * 2);
  float* tg       = (float*)alloc((size_t)B_ * HID_ * 4);
  float* preact   = (float*)alloc((size_t)B_ * NN_ * 4);
  float* hist     = (float*)alloc((size_t)B_ * NN_ * HIST_ * 4);
  float* Aacts    = (float*)alloc((size_t)B_ * TICKS_ * NN_ * 4);
  float* zproj    = (float*)alloc((size_t)B_ * HID_ * 4);

  float* logits   = (float*)d_out;
  float* sync_out = logits + (size_t)B_ * VOCAB_;
  float* cert_out = sync_out + B_;

  // --- BitNet quantization of the 4 BitLinear weights ---
  const float* bw[4]  = { ip_w, s1_w, s2_w, np_w };
  const int    bc[4]  = { HID_ * 2 * HID_, HID_ * 2 * HID_, HID_ * HID_, HID_ * NN_ };
  bf16*        bq[4]  = { ipq, s1q, s2q, npq };
  for (int i = 0; i < 4; i++) {
    partial_absclip_kernel<<<256, 256, 0, stream>>>(bw[i], bc[i], partials);
    finalize_gamma_kernel<<<1, 256, 0, stream>>>(partials, bc[i], gam + i);
    quantize_bf16_kernel<<<256, 256, 0, stream>>>(bw[i], bc[i], gam + i, bq[i]);
  }
  cvt_bf16_kernel<<<128, 256, 0, stream>>>(zp_w, HID_ * NN_, zpwb);
  cvt_bf16_kernel<<<128, 256, 0, stream>>>(s3_w, NN_ * HID_, s3wb);
  pack_fused_kernel<<<256, 256, 0, stream>>>(l0, l1, fusedb);
  init_z_kernel<<<64, 256, 0, stream>>>(init_z, zbf);
  init_hist_kernel<<<512, 256, 0, stream>>>(init_h, hist);

  // --- att = LN(BitLinear(fused)); bf16 copy straight into hcat[:,512:] ---
  gemm_bf16_kernel<<<dim3(8, 16), 128, 0, stream>>>(fusedb, ipq, ip_b, tg, HID_, 2 * HID_);
  ln_kernel<<<B_, 256, 0, stream>>>(tg, ip_ln_g, ip_ln_b, 0, (float*)nullptr, 0, hcatb + HID_, 2 * HID_);

  // --- tick loop ---
  for (int tick = 0; tick < TICKS_; tick++) {
    gemm_bf16_kernel<<<dim3(8, 16), 128, 0, stream>>>(zbf, zpwb, zp_b, tg, HID_, NN_);
    ln_kernel<<<B_, 256, 0, stream>>>(tg, zn_g, zn_b, 0, (float*)nullptr, 0, hcatb, 2 * HID_);
    gemm_bf16_kernel<<<dim3(8, 16), 128, 0, stream>>>(hcatb, s1q, s1_b, tg, HID_, 2 * HID_);
    ln_kernel<<<B_, 256, 0, stream>>>(tg, ln1_g, ln1_b, 1, (float*)nullptr, 0, h1b, HID_);
    gemm_bf16_kernel<<<dim3(8, 16), 128, 0, stream>>>(h1b, s2q, s2_b, tg, HID_, HID_);
    ln_kernel<<<B_, 256, 0, stream>>>(tg, ln2_g, ln2_b, 1, (float*)nullptr, 0, h2b, HID_);
    gemm_bf16_kernel<<<dim3(4, 16), 128, 0, stream>>>(h2b, s3wb, s3_b, preact, NN_, HID_);
    nlm_kernel<<<B_, 256, 0, stream>>>(preact, hist, n1_w, n1_b, n2_w, n2_b, Aacts, tick, zbf);
  }

  // --- aggregation, projection, heads ---
  agg_kernel<<<B_, 256, 0, stream>>>(Aacts, zaggb, sync_out);
  gemm_bf16_kernel<<<dim3(8, 16), 128, 0, stream>>>(zaggb, npq, np_b, tg, HID_, NN_);
  ln_kernel<<<B_, 256, 0, stream>>>(tg, np_ln_g, np_ln_b, 0, zproj, HID_, zprojb, HID_);
  gemm_vocab_kernel<<<dim3(VOCAB_ / 128), 256, 0, stream>>>(zprojb, out_w, out_b, logits, VOCAB_, HID_);
  cert_kernel<<<B_, 256, 0, stream>>>(zproj, cert_w, cert_b, cert_out);
}
